// RQVAE_36550171689071
// MI455X (gfx1250) — matrix operand order repaired and checked by reading, not compiled, not run on hardware
//
#include <hip/hip_runtime.h>
#include <hip/hip_bf16.h>

// RQ-VAE forward for MI455X (gfx1250, wave32).
// GEMM per level runs on v_wmma_f32_16x16x32_bf16 (f32 accumulate); everything
// else (softmax, argmin/argmax, residual update, loss) is fused per 16-row tile.

typedef __bf16 bf16_t;
typedef __attribute__((ext_vector_type(16))) __bf16 v16bf;
typedef __attribute__((ext_vector_type(8)))  __bf16 v8bf;
typedef __attribute__((ext_vector_type(8)))  float  v8f;

#define B_ROWS 16384
#define DIM    512
#define KCB    256
#define NLV    4
#define ASTRIDE 536   // bf16 elems per LDS A row: 16B-aligned, bank-staggered
#define SSTRIDE 260   // f32 elems per LDS score row

__device__ __forceinline__ float wave_sum(float v) {
  #pragma unroll
  for (int off = 16; off; off >>= 1) v += __shfl_down(v, off);
  return __shfl(v, 0);
}

// Convert rows of f32 matrix to bf16 and compute per-row squared norms.
// One wave per row (8 rows per 256-thread block).
__global__ __launch_bounds__(256)
void prep_rows_kernel(const float* __restrict__ src, bf16_t* __restrict__ dst,
                      float* __restrict__ norm) {
  const int row  = blockIdx.x * 8 + (threadIdx.x >> 5);
  const int lane = threadIdx.x & 31;
  const float* s = src + (size_t)row * DIM;
  float acc = 0.f;
  for (int d = lane; d < DIM; d += 32) {
    float v = s[d];
    acc += v * v;
    dst[(size_t)row * DIM + d] = (bf16_t)v;
  }
  acc = wave_sum(acc);
  if (lane == 0) norm[row] = acc;
}

// One block: 16 rows x 256 codewords. 16 waves; wave w computes score
// columns [16w,16w+16) for all 16 rows via WMMA, then wave r reduces row r.
__global__ __launch_bounds__(512)
void vq_level_kernel(const float* __restrict__ resid_in,
                     float*       __restrict__ resid_out,
                     bf16_t*      __restrict__ Rbf,     // in/out (next level)
                     float*       __restrict__ rnorm,   // in/out (next level)
                     const float* __restrict__ W,       // f32 codebook [256][512]
                     const bf16_t* __restrict__ Wbf,    // bf16 codebook [256][512]
                     const float* __restrict__ wnorm,   // [256]
                     float* __restrict__ quant_out,     // [B][512]
                     float* __restrict__ idx_out,       // [B][4] (as float)
                     float* __restrict__ probs_out,     // [B][256][4]
                     float* __restrict__ loss_out,      // [B]
                     int lvl) {
  __shared__ bf16_t Alds[16 * ASTRIDE];
  __shared__ float  Stile[16 * SSTRIDE];
  __shared__ float  rnS[16];

  const int tid  = threadIdx.x;
  const int wv   = tid >> 5;
  const int lane = tid & 31;
  const int b0   = blockIdx.x * 16;

  // ---- Phase 1: stage the 16x512 bf16 residual tile + row norms in LDS ----
  for (int c = tid; c < 1024; c += 512) {          // 1024 chunks of 8 bf16
    int row = c >> 6, d = (c & 63) << 3;
    *(v8bf*)&Alds[row * ASTRIDE + d] =
        *(const v8bf*)&Rbf[(size_t)(b0 + row) * DIM + d];
  }
  if (tid < 16) rnS[tid] = rnorm[b0 + tid];
  __syncthreads();

  // ---- Phase 2: S = R * W^T via 16 chained bf16 WMMAs over D=512 ----
  const int h  = lane >> 4;     // half-wave select (ISA 16-bit A/B layouts)
  const int al = lane & 15;     // A row within tile / B column within tile
  const int cw = wv * 16 + al;  // global codeword for this lane's B column
  const bf16_t* arow = &Alds[al * ASTRIDE];
  const bf16_t* brow = &Wbf[(size_t)cw * DIM];
  v8f acc = {};
  #pragma unroll
  for (int d0 = 0; d0 < DIM; d0 += 32) {
    // A 16x32 bf16: elems 0..7 = K d0+h*8.., elems 8..15 = K d0+16+h*8..
    v8bf a0 = *(const v8bf*)&arow[d0 + h * 8];
    v8bf a1 = *(const v8bf*)&arow[d0 + 16 + h * 8];
    // B 32x16 bf16: lane col = al, elems 0..7 = K d0+h*8.., elems 8..15 = K d0+16+h*8..
    v8bf bb0 = *(const v8bf*)&brow[d0 + h * 8];
    v8bf bb1 = *(const v8bf*)&brow[d0 + 16 + h * 8];
    v16bf a, b;
    #pragma unroll
    for (int e = 0; e < 8; ++e) {
      a[e] = a0[e]; a[e + 8] = a1[e];
      b[e] = bb0[e]; b[e + 8] = bb1[e];
    }
    acc = __builtin_amdgcn_wmma_f32_16x16x32_bf16(
        false, a, false, b, (short)0, acc, false, false);
  }
  // d2 = ||r||^2 + ||w||^2 - 2*dot ; C layout: lane<16 -> M=j, lane>=16 -> M=j+8
  const float wn = wnorm[cw];
  #pragma unroll
  for (int j = 0; j < 8; ++j) {
    int rr = j + 8 * h;
    Stile[rr * SSTRIDE + cw] = rnS[rr] + wn - 2.0f * acc[j];
  }
  __syncthreads();

  // ---- Phase 3: wave r reduces row r (min/argmin, max/argmax, probs) ----
  const int r = wv;
  const int b = b0 + r;
  const float* Srow = &Stile[r * SSTRIDE];
  float mn = 3.0e38f;  int mnI = 0;
  float mx = -3.0e38f; int mxI = 0;
  #pragma unroll
  for (int j = 0; j < 8; ++j) {
    int c = j * 32 + lane;
    float v = Srow[c];
    if (v < mn || (v == mn && c < mnI)) { mn = v; mnI = c; }
    if (v > mx || (v == mx && c < mxI)) { mx = v; mxI = c; }
  }
  #pragma unroll
  for (int off = 16; off; off >>= 1) {           // first-index tie-break
    float v = __shfl_down(mn, off); int i = __shfl_down(mnI, off);
    if (v < mn || (v == mn && i < mnI)) { mn = v; mnI = i; }
    v = __shfl_down(mx, off); i = __shfl_down(mxI, off);
    if (v > mx || (v == mx && i < mxI)) { mx = v; mxI = i; }
  }
  mn = __shfl(mn, 0); mnI = __shfl(mnI, 0);
  mx = __shfl(mx, 0); mxI = __shfl(mxI, 0);

  int sel;
  float* prow = &probs_out[(size_t)b * KCB * NLV + lvl];
  if (lvl < 3) {
    // softmax(-d2) = exp(min-d2)/sum ; argmax(probs) == argmin(d2)
    float s = 0.f;
    #pragma unroll
    for (int j = 0; j < 8; ++j) s += __expf(mn - Srow[j * 32 + lane]);
    s = wave_sum(s);
    float inv = 1.0f / s;
    #pragma unroll
    for (int j = 0; j < 8; ++j) {
      int c = j * 32 + lane;
      prow[(size_t)c * NLV] = __expf(mn - Srow[c]) * inv;
    }
    sel = mnI;
  } else {
    // Sinkhorn on `normed` collapses to a scale of B/(K*B)^30 = 2^-646;
    // the result underflows f32 to +/-0, and argmax(normed) == argmax(d2).
    const double SCALE3 = __builtin_bit_cast(double, 0x1790000000000000ULL); // 2^-646
    float mid = 0.5f * (mx + mn);
    float inv = 1.0f / (mx - mid + 1e-8f);
    #pragma unroll
    for (int j = 0; j < 8; ++j) {
      int c = j * 32 + lane;
      float nrm = (Srow[c] - mid) * inv;
      prow[(size_t)c * NLV] = (float)((double)nrm * SCALE3);
    }
    sel = mxI;
  }
  if (lane == 0) idx_out[(size_t)b * NLV + lvl] = (float)sel;

  // ---- Phase 4: residual update, quant accumulate, loss, next-level prep ----
  const float* Wr = &W[(size_t)sel * DIM];
  float lossAcc = 0.f, rnAcc = 0.f;
  for (int d = lane; d < DIM; d += 32) {
    float q  = Wr[d];
    float ro = resid_in[(size_t)b * DIM + d];
    float df = q - ro;
    lossAcc += df * df;
    float rn2 = ro - q;
    resid_out[(size_t)b * DIM + d] = rn2;
    rnAcc += rn2 * rn2;
    Rbf[(size_t)b * DIM + d] = (bf16_t)rn2;
    if (lvl == 0) quant_out[(size_t)b * DIM + d] = q;
    else          quant_out[(size_t)b * DIM + d] += q;
  }
  lossAcc = wave_sum(lossAcc);
  rnAcc   = wave_sum(rnAcc);
  if (lane == 0) {
    rnorm[b] = rnAcc;
    // loss_i = (1+MU)*||W[sel]-r||^2/DIM ; final = mean over 4 levels
    float contrib = lossAcc * (1.25f / (float)(DIM * NLV));
    if (lvl == 0) loss_out[b] = contrib;
    else          loss_out[b] += contrib;
  }
}

extern "C" void kernel_launch(void* const* d_in, const int* in_sizes, int n_in,
                              void* d_out, int out_size, void* d_ws, size_t ws_size,
                              hipStream_t stream) {
  (void)in_sizes; (void)n_in; (void)out_size; (void)ws_size;
  const float* x  = (const float*)d_in[0];                 // [16384][512]
  const float* cb = (const float*)d_in[1];                 // [4][256][512]

  uint8_t* ws = (uint8_t*)d_ws;
  size_t off = 0;
  float*  resid = (float*)(ws + off);  off += (size_t)B_ROWS * DIM * sizeof(float);
  bf16_t* Rbf   = (bf16_t*)(ws + off); off += (size_t)B_ROWS * DIM * sizeof(bf16_t);
  bf16_t* Wbf   = (bf16_t*)(ws + off); off += (size_t)NLV * KCB * DIM * sizeof(bf16_t);
  float*  rnorm = (float*)(ws + off);  off += (size_t)B_ROWS * sizeof(float);
  float*  wnorm = (float*)(ws + off);  off += (size_t)NLV * KCB * sizeof(float);

  float* quant = (float*)d_out;                            // [B][512]
  float* idxo  = quant + (size_t)B_ROWS * DIM;             // [B][4]
  float* probs = idxo + (size_t)B_ROWS * NLV;              // [B][256][4]
  float* losso = probs + (size_t)B_ROWS * KCB * NLV;       // [B]

  // bf16 codebooks + ||W||^2 (all 4 levels), bf16 x + ||x||^2
  prep_rows_kernel<<<(NLV * KCB) / 8, 256, 0, stream>>>(cb, Wbf, wnorm);
  prep_rows_kernel<<<B_ROWS / 8, 256, 0, stream>>>(x, Rbf, rnorm);

  for (int l = 0; l < NLV; ++l) {
    const float* rin = (l == 0) ? x : resid;   // never mutate d_in
    vq_level_kernel<<<B_ROWS / 16, 512, 0, stream>>>(
        rin, resid, Rbf, rnorm,
        cb + (size_t)l * KCB * DIM, Wbf + (size_t)l * KCB * DIM, wnorm + l * KCB,
        quant, idxo, probs, losso, l);
  }
}
